// Block_Atom_18090402250769
// MI455X (gfx1250) — matrix operand order repaired and checked by reading, not compiled, not run on hardware
//
#include <hip/hip_runtime.h>
#include <hip/hip_bf16.h>

typedef __attribute__((ext_vector_type(16))) _Float16 v16h;
typedef __attribute__((ext_vector_type(8)))  float    v8f;

namespace {
constexpr int B_   = 8;
constexpr int NA   = 4096;
constexpr int NAA  = 1024;
constexpr int NPC  = 4096;
constexpr int K_   = 16;
constexpr int KNC  = 14;
constexpr int G_   = 32;
constexpr int DA   = 12;
constexpr int D_   = 128;

// workspace byte offsets
constexpr size_t OFF_WNEM   = 0;                       // 24576 u32 = 98304 B
constexpr size_t OFF_WATT   = 98304;                   // 8192 u32  = 32768 B
constexpr size_t OFF_WFEAT  = 131072;                  // 8192 u32  = 32768 B
constexpr size_t OFF_MASKY  = 163840;                  // B*NA f32  = 131072 B
constexpr size_t OFF_ATT    = 294912;                  // B*NA*128 f32 = 16 MB
constexpr size_t OFF_FEAT   = 17072128;                // 16 MB
constexpr size_t OFF_POOLED = 33849344;                // B*NAA*128 f32 = 4 MB
constexpr size_t OFF_SUMS   = 38043648;                // 257 f32 (S[128],Sq[128],Nm)
}

union U16h  { unsigned int u[8]; v16h v; };
union HPack { _Float16 h[2]; unsigned int u; };

// ---------------------------------------------------------------- prep ------
__global__ void prep_kernel(const float* __restrict__ wnem,
                            const float* __restrict__ watt,
                            const float* __restrict__ wfeat,
                            unsigned int* __restrict__ wnem_pack,
                            unsigned int* __restrict__ watt_pack,
                            unsigned int* __restrict__ wfeat_pack,
                            float* __restrict__ sums) {
  int t = blockIdx.x * blockDim.x + threadIdx.x;
  if (t < 24576) {                       // W_nem: 8 ntiles * 12 kk chunks
    int v = t & 7, lane = (t >> 3) & 31, rest = t >> 8;
    int kk = rest % 12, nt = rest / 12;
    int N  = nt * 16 + (lane & 15);
    int Kb = kk * 32 + 8 * (lane >> 4) + 2 * (v & 3) + 16 * (v >> 2);
    HPack pk;
    pk.h[0] = (_Float16)wnem[Kb * 128 + N];
    pk.h[1] = (_Float16)wnem[(Kb + 1) * 128 + N];
    wnem_pack[t] = pk.u;
  } else if (t < 32768) {                // W_att: 8 ntiles * 4 kk chunks
    int u2 = t - 24576;
    int v = u2 & 7, lane = (u2 >> 3) & 31, rest = u2 >> 8;
    int kk = rest & 3, nt = rest >> 2;
    int N  = nt * 16 + (lane & 15);
    int Kb = kk * 32 + 8 * (lane >> 4) + 2 * (v & 3) + 16 * (v >> 2);
    HPack pk;
    pk.h[0] = (_Float16)watt[Kb * 128 + N];
    pk.h[1] = (_Float16)watt[(Kb + 1) * 128 + N];
    watt_pack[u2] = pk.u;
  } else if (t < 40960) {                // W_feat
    int u3 = t - 32768;
    int v = u3 & 7, lane = (u3 >> 3) & 31, rest = u3 >> 8;
    int kk = rest & 3, nt = rest >> 2;
    int N  = nt * 16 + (lane & 15);
    int Kb = kk * 32 + 8 * (lane >> 4) + 2 * (v & 3) + 16 * (v >> 2);
    HPack pk;
    pk.h[0] = (_Float16)wfeat[Kb * 128 + N];
    pk.h[1] = (_Float16)wfeat[(Kb + 1) * 128 + N];
    wfeat_pack[u3] = pk.u;
  } else if (t < 41217) {                // zero BN accumulators
    sums[t - 40960] = 0.0f;
  }
}

// ------------------------------------------------------------- atom ---------
// 1 block = 16 atoms, 128 threads = 4 waves.
__global__ void __launch_bounds__(128)
atom_kernel(const float* __restrict__ point_clouds,
            const float* __restrict__ mask_atom,
            const float* __restrict__ attr_table,
            const float* __restrict__ gauss_centers,
            const int*   __restrict__ frame_idx,
            const int*   __restrict__ attr_idx,
            const int*   __restrict__ nb_idx,
            const unsigned int* __restrict__ wnem_pack,
            const unsigned int* __restrict__ watt_pack,
            const unsigned int* __restrict__ wfeat_pack,
            float* __restrict__ maskY,
            float* __restrict__ att_out,
            float* __restrict__ feat_out) {
  __shared__ float sCenter[16][3];
  __shared__ float sR[16][9];
  __shared__ float sMask[16];
  __shared__ float sGC[32][3];
  __shared__ float sRel[16][16][3];
  __shared__ float sAnb[16][16][12];
  __shared__ unsigned int sAFrag[12 * 32 * 8];   // outer 16x384 as f16 A-fragments
  __shared__ unsigned int sYFrag[4 * 32 * 8];    // y     16x128 as f16 A-fragments

  const int tid  = threadIdx.x;
  const int base = blockIdx.x * 16;
  const int b    = base / NA;
  const float* pc = point_clouds + (size_t)b * NPC * 3;

  // ---- per-atom frame, attrs, mask ----
  if (tid < 16) {
    int gRow = base + tid;
    int f0 = frame_idx[gRow * 3 + 0];
    int f1 = frame_idx[gRow * 3 + 1];
    int f2 = frame_idx[gRow * 3 + 2];
    float p0x = pc[f0*3], p0y = pc[f0*3+1], p0z = pc[f0*3+2];
    float cx  = pc[f1*3], cy  = pc[f1*3+1], cz  = pc[f1*3+2];
    float p2x = pc[f2*3], p2y = pc[f2*3+1], p2z = pc[f2*3+2];
    float ax = p2x - cx, ay = p2y - cy, az = p2z - cz;
    float an = sqrtf(ax*ax + ay*ay + az*az) + 1e-8f;
    float u1x = ax/an, u1y = ay/an, u1z = az/an;
    float vx = p0x - cx, vy = p0y - cy, vz = p0z - cz;
    float pr = vx*u1x + vy*u1y + vz*u1z;
    float wx = vx - pr*u1x, wy = vy - pr*u1y, wz = vz - pr*u1z;
    float wn = sqrtf(wx*wx + wy*wy + wz*wz) + 1e-8f;
    float u2x = wx/wn, u2y = wy/wn, u2z = wz/wn;
    float u3x = u1y*u2z - u1z*u2y;
    float u3y = u1z*u2x - u1x*u2z;
    float u3z = u1x*u2y - u1y*u2x;
    sCenter[tid][0] = cx; sCenter[tid][1] = cy; sCenter[tid][2] = cz;
    sR[tid][0] = u1x; sR[tid][1] = u1y; sR[tid][2] = u1z;
    sR[tid][3] = u2x; sR[tid][4] = u2y; sR[tid][5] = u2z;
    sR[tid][6] = u3x; sR[tid][7] = u3y; sR[tid][8] = u3z;
    int ai = attr_idx[gRow];
    float anyNZ = 0.0f;
    #pragma unroll
    for (int a = 0; a < DA; ++a)
      if (attr_table[ai * DA + a] != 0.0f) anyNZ = 1.0f;
    float my = mask_atom[gRow] * anyNZ;
    sMask[tid]  = my;
    maskY[gRow] = my;
  }
  if (tid < 96) sGC[tid / 3][tid % 3] = gauss_centers[tid];
  __syncthreads();

  // ---- neighbor gather: rel coords + masked attrs ----
  #pragma unroll
  for (int q = 0; q < 2; ++q) {
    int id = tid + q * 128;                // (m,k) pairs: 256
    int m = id >> 4, k = id & 15;
    int gRow = base + m;
    int j  = nb_idx[(size_t)gRow * K_ + k];
    int jb = b * NA + j;
    int fc = frame_idx[jb * 3 + 1];
    float dx = pc[fc*3]   - sCenter[m][0];
    float dy = pc[fc*3+1] - sCenter[m][1];
    float dz = pc[fc*3+2] - sCenter[m][2];
    sRel[m][k][0] = sR[m][0]*dx + sR[m][1]*dy + sR[m][2]*dz;
    sRel[m][k][1] = sR[m][3]*dx + sR[m][4]*dy + sR[m][5]*dz;
    sRel[m][k][2] = sR[m][6]*dx + sR[m][7]*dy + sR[m][8]*dz;
    int ai = attr_idx[jb];
    float av[DA]; float anyNZ = 0.0f;
    #pragma unroll
    for (int a = 0; a < DA; ++a) {
      av[a] = attr_table[ai * DA + a];
      if (av[a] != 0.0f) anyNZ = 1.0f;
    }
    #pragma unroll
    for (int a = 0; a < DA; ++a) sAnb[m][k][a] = av[a] * anyNZ;
  }
  __syncthreads();

  // ---- outer[g][a] = sum_k exp(-0.5*d2)*a_nb, stored as A-fragments (f16) ----
  _Float16* afh = (_Float16*)sAFrag;
  #pragma unroll
  for (int q = 0; q < 4; ++q) {
    int id = tid + q * 128;                // (m,g) pairs: 512
    int m = id >> 5, g = id & 31;
    float gx = sGC[g][0], gy = sGC[g][1], gz = sGC[g][2];
    float acc[DA];
    #pragma unroll
    for (int a = 0; a < DA; ++a) acc[a] = 0.0f;
    #pragma unroll
    for (int k = 0; k < K_; ++k) {
      float rx = sRel[m][k][0] - gx;
      float ry = sRel[m][k][1] - gy;
      float rz = sRel[m][k][2] - gz;
      float gv = __expf(-0.5f * (rx*rx + ry*ry + rz*rz));
      #pragma unroll
      for (int a = 0; a < DA; ++a) acc[a] += gv * sAnb[m][k][a];
    }
    #pragma unroll
    for (int a = 0; a < DA; ++a) {
      int Kt = g * DA + a;                 // K index in [0,384)
      int kk = Kt >> 5, kc = Kt & 31;
      int tl = m + 16 * ((kc >> 3) & 1);
      int v2 = (kc < 16 ? 0 : 4) + ((kc & 7) >> 1);
      int p  = kc & 1;
      afh[((kk * 32 + tl) * 8 + v2) * 2 + p] = (_Float16)acc[a];
    }
  }
  __syncthreads();

  // ---- WMMA GEMM 1: y(16x128) = outer(16x384) x W_nem(384x128) ----
  const int wv = tid >> 5, ln = tid & 31;
  const int lh = ln >> 4, lm = ln & 15;
  _Float16* yfh = (_Float16*)sYFrag;

  for (int ti = 0; ti < 2; ++ti) {
    int nt = wv + 4 * ti;                  // N-tile 0..7
    v8f acc = {};
    for (int kk = 0; kk < 12; ++kk) {
      U16h A, Bf;
      const unsigned int* ap = sAFrag + (kk * 32 + ln) * 8;
      const unsigned int* bp = wnem_pack + ((size_t)(nt * 12 + kk) * 32 + ln) * 8;
      #pragma unroll
      for (int vv = 0; vv < 8; ++vv) { A.u[vv] = ap[vv]; Bf.u[vv] = bp[vv]; }
      acc = __builtin_amdgcn_wmma_f32_16x16x32_f16(
          false, A.v, false, Bf.v, (short)0, acc, false, false);
    }
    #pragma unroll
    for (int r = 0; r < 8; ++r) {
      int M = r + 8 * lh;
      float val = acc[r] * sMask[M] * (1.0f / (float)K_);   // y = y*mask/K
      int c   = nt * 16 + lm;                                // feature col
      int kk2 = c >> 5, kc = c & 31;
      int tl  = M + 16 * ((kc >> 3) & 1);
      int v2  = (kc < 16 ? 0 : 4) + ((kc & 7) >> 1);
      int p   = kc & 1;
      yfh[((kk2 * 32 + tl) * 8 + v2) * 2 + p] = (_Float16)val;
    }
  }
  __syncthreads();

  // ---- WMMA GEMM 2/3: att = y @ W_att * mask ; feat = y @ W_feat * mask ----
  for (int q = 0; q < 4; ++q) {
    int tt  = wv + 4 * q;                  // 16 tiles: [0..7]=att, [8..15]=feat
    int mat = tt >> 3, nt2 = tt & 7;
    const unsigned int* wpk = mat ? wfeat_pack : watt_pack;
    float* dst = mat ? feat_out : att_out;
    v8f acc = {};
    for (int kk2 = 0; kk2 < 4; ++kk2) {
      U16h A, Bf;
      const unsigned int* ap = sYFrag + (kk2 * 32 + ln) * 8;
      const unsigned int* bp = wpk + ((size_t)(nt2 * 4 + kk2) * 32 + ln) * 8;
      #pragma unroll
      for (int vv = 0; vv < 8; ++vv) { A.u[vv] = ap[vv]; Bf.u[vv] = bp[vv]; }
      acc = __builtin_amdgcn_wmma_f32_16x16x32_f16(
          false, A.v, false, Bf.v, (short)0, acc, false, false);
    }
    #pragma unroll
    for (int r = 0; r < 8; ++r) {
      int M = r + 8 * lh;
      float val = acc[r] * sMask[M];
      dst[(size_t)(base + M) * D_ + nt2 * 16 + lm] = val;
    }
  }
}

// ------------------------------------------------------------- pooling ------
__global__ void __launch_bounds__(128)
pool_kernel(const float* __restrict__ att,
            const float* __restrict__ feat,
            const float* __restrict__ maskY,
            const float* __restrict__ mask_aa,
            const int*   __restrict__ seq_atom,
            const int*   __restrict__ seq_aa,
            const int*   __restrict__ aa_nb,
            float* __restrict__ pooled,
            float* __restrict__ sums) {
  int bi = blockIdx.x;                 // (b, i) flattened over B*NAA
  int b  = bi / NAA;
  int d  = threadIdx.x;
  __shared__ int   sJ[KNC];
  __shared__ float sGate[KNC];
  __shared__ float sM[KNC];
  if (d < KNC) {
    int j = aa_nb[(size_t)bi * KNC + d];
    int jb = b * NA + j;
    sJ[d] = jb;
    int diff = seq_atom[jb] - seq_aa[bi];
    if (diff < 0) diff = -diff;
    if (diff > 1) diff = 1;
    sGate[d] = 1.0f - (float)diff;
    sM[d]    = maskY[jb];
  }
  __syncthreads();

  float lg[KNC];
  float mx = -1e30f;
  #pragma unroll
  for (int k = 0; k < KNC; ++k) {
    float al = (sM[k] > 0.0f) ? att[(size_t)sJ[k] * D_ + d] : -1e9f;
    lg[k] = al;
    mx = fmaxf(mx, al);
  }
  float s = 0.0f;
  #pragma unroll
  for (int k = 0; k < KNC; ++k) { lg[k] = __expf(lg[k] - mx); s += lg[k]; }
  float inv = 1.0f / s;
  float wsum = 0.0f;
  #pragma unroll
  for (int k = 0; k < KNC; ++k) { lg[k] = lg[k] * inv * sGate[k] * sM[k]; wsum += lg[k]; }
  float wn = 1.0f / (wsum + 1e-8f);
  float p = 0.0f;
  #pragma unroll
  for (int k = 0; k < KNC; ++k) p += lg[k] * wn * feat[(size_t)sJ[k] * D_ + d];
  float ma = mask_aa[bi];
  p *= ma;
  pooled[(size_t)bi * D_ + d] = p;
  atomicAdd(&sums[d],        p * ma);
  atomicAdd(&sums[128 + d],  p * p * ma);
  if (d == 0) atomicAdd(&sums[256], ma);
}

// ------------------------------------------------------------- batchnorm ----
__global__ void __launch_bounds__(128)
bn_kernel(const float* __restrict__ pooled,
          const float* __restrict__ sums,
          const float* __restrict__ gamma,
          const float* __restrict__ beta,
          const float* __restrict__ mask_aa,
          float* __restrict__ out) {
  int bi = blockIdx.x;
  int d  = threadIdx.x;
  float Nm = sums[256];
  float n  = Nm + 1e-8f;
  float S  = sums[d];
  float Sq = sums[128 + d];
  float mean = S / n;
  float var  = (Sq - 2.0f * mean * S + mean * mean * Nm) / n;
  float p = pooled[(size_t)bi * D_ + d];
  float o = gamma[d] * (p - mean) * rsqrtf(var + 1e-5f) + beta[d];
  float ma = mask_aa[bi];
  o = fmaxf(o * ma, 0.0f);
  out[(size_t)bi * D_ + d] = o;
  if (d == 0) out[(size_t)B_ * NAA * D_ + bi] = ma;   // mask_aa tuple tail
}

// ------------------------------------------------------------- launcher -----
extern "C" void kernel_launch(void* const* d_in, const int* in_sizes, int n_in,
                              void* d_out, int out_size, void* d_ws, size_t ws_size,
                              hipStream_t stream) {
  const float* point_clouds = (const float*)d_in[0];
  const float* mask_atom    = (const float*)d_in[1];
  const float* mask_aa      = (const float*)d_in[2];
  const float* attr_table   = (const float*)d_in[3];
  const float* gauss_ctr    = (const float*)d_in[4];
  const float* W_nem        = (const float*)d_in[5];
  const float* W_att        = (const float*)d_in[6];
  const float* W_feat       = (const float*)d_in[7];
  const float* bn_gamma     = (const float*)d_in[8];
  const float* bn_beta      = (const float*)d_in[9];
  const int*   frame_idx    = (const int*)d_in[10];
  const int*   attr_idx     = (const int*)d_in[11];
  const int*   nb_idx       = (const int*)d_in[12];
  const int*   seq_atom     = (const int*)d_in[13];
  const int*   seq_aa       = (const int*)d_in[14];
  const int*   aa_nb        = (const int*)d_in[15];

  char* ws = (char*)d_ws;
  unsigned int* wnem_pack  = (unsigned int*)(ws + OFF_WNEM);
  unsigned int* watt_pack  = (unsigned int*)(ws + OFF_WATT);
  unsigned int* wfeat_pack = (unsigned int*)(ws + OFF_WFEAT);
  float* maskY   = (float*)(ws + OFF_MASKY);
  float* att_ws  = (float*)(ws + OFF_ATT);
  float* feat_ws = (float*)(ws + OFF_FEAT);
  float* pooled  = (float*)(ws + OFF_POOLED);
  float* sums    = (float*)(ws + OFF_SUMS);

  float* out = (float*)d_out;

  prep_kernel<<<162, 256, 0, stream>>>(W_nem, W_att, W_feat,
                                       wnem_pack, watt_pack, wfeat_pack, sums);

  atom_kernel<<<(B_ * NA) / 16, 128, 0, stream>>>(
      point_clouds, mask_atom, attr_table, gauss_ctr,
      frame_idx, attr_idx, nb_idx,
      wnem_pack, watt_pack, wfeat_pack,
      maskY, att_ws, feat_ws);

  pool_kernel<<<B_ * NAA, 128, 0, stream>>>(
      att_ws, feat_ws, maskY, mask_aa, seq_atom, seq_aa, aa_nb,
      pooled, sums);

  bn_kernel<<<B_ * NAA, 128, 0, stream>>>(
      pooled, sums, bn_gamma, bn_beta, mask_aa, out);
}
